// PNA_10703058502292
// MI455X (gfx1250) — matrix operand order repaired and checked by reading, compiled, not run on hardware
//
#include <hip/hip_runtime.h>
#include <hip/hip_bf16.h>
#include <stdint.h>

// Problem constants (match reference)
#define NN 40000
#define GG 400
#define EE 600000

// ---------- CDNA5 WMMA types ----------
typedef __attribute__((ext_vector_type(16))) __bf16   v16bf;
typedef __attribute__((ext_vector_type(8)))  float    v8f;
typedef unsigned u32x4 __attribute__((ext_vector_type(4)));

__device__ __forceinline__ unsigned short f2bf(float f) {
  unsigned u = __float_as_uint(f);
  u += 0x7fffu + ((u >> 16) & 1u);           // round-to-nearest-even
  return (unsigned short)(u >> 16);
}
// order-preserving float -> u32 key (for atomicMin/Max on floats)
__device__ __forceinline__ unsigned fkey(float f) {
  unsigned u = __float_as_uint(f);
  return u ^ ((unsigned)((int)u >> 31) | 0x80000000u);
}
__device__ __forceinline__ float kdec(unsigned k) {
  unsigned u = (k & 0x80000000u) ? (k ^ 0x80000000u) : ~k;
  return __uint_as_float(u);
}

// ---------- generic fills ----------
__global__ void fill_f32_kernel(float* p, float v, int n) {
  int t = blockIdx.x * 256 + threadIdx.x; if (t < n) p[t] = v;
}
__global__ void fill_u32_kernel(unsigned* p, unsigned v, int n) {
  int t = blockIdx.x * 256 + threadIdx.x; if (t < n) p[t] = v;
}

// ---------- degree / avg-log-degree ----------
__global__ void deg_kernel(const int* __restrict__ dst, float* __restrict__ deg, int n) {
  int t = blockIdx.x * 256 + threadIdx.x; if (t < n) atomicAdd(&deg[dst[t]], 1.0f);
}
__global__ void logsum_kernel(const float* __restrict__ deg, float* __restrict__ scal, int n) {
  int t = blockIdx.x * 256 + threadIdx.x; if (t < n) atomicAdd(&scal[0], logf(deg[t] + 1.0f));
}
__global__ void avg_kernel(float* scal) { scal[1] = scal[0] / (float)NN; }

// ---------- layer-0 input conversion (fp32 [N,15] -> bf16 [N,32], zero-padded) ----------
__global__ void convert_x_kernel(const float* __restrict__ x, unsigned short* __restrict__ xb, int total) {
  int t = blockIdx.x * 256 + threadIdx.x; if (t >= total) return;
  int n = t / 15, c = t - n * 15;
  xb[n * 32 + c] = f2bf(x[t]);
}

// ---------- fuse edge-encoder into pre-NN:  Wfe = eeW @ preW[2fi:], bfe = eeb @ preW[2fi:] + preb
__global__ void fuse_edge_kernel(const float* __restrict__ eeW, const float* __restrict__ eeb,
                                 const float* __restrict__ preW, const float* __restrict__ preb,
                                 int fi, float* __restrict__ Wfe, float* __restrict__ bfe) {
  int t = blockIdx.x * 256 + threadIdx.x; if (t >= 4 * fi) return;
  int r = t / fi, c = t - r * fi;
  if (r < 3) {
    float s = 0.f;
    for (int k = 0; k < fi; ++k) s += eeW[r * fi + k] * preW[(size_t)(2 * fi + k) * fi + c];
    Wfe[r * fi + c] = s;
  } else {
    float s = preb[c];
    for (int k = 0; k < fi; ++k) s += eeb[k] * preW[(size_t)(2 * fi + k) * fi + c];
    bfe[c] = s;
  }
}

// ---------- pack weights into CDNA5 B-fragment layout ----------
// k order chosen here (A loads match): element i (0..15) of lane L holds K = 16*(L>>4)+i, N = L&15.
// Packed as [kt][nt][lane][i] contiguous ushorts -> GEMM does 2x b128 loads/lane,
// and consecutive nt slabs are contiguous, so an NT-wide register block reads
// one contiguous 512*NT-ushort slab per K-step.
__global__ void pack_b_plain(const float* __restrict__ W, int Ksrc, int Nsrc,
                             unsigned short* __restrict__ Bp, int ntn, int total) {
  int t = blockIdx.x * 256 + threadIdx.x; if (t >= total) return;
  int i = t & 15, lane = (t >> 4) & 31, tile = t >> 9;
  int kt = tile / ntn, nt = tile - kt * ntn;
  int K = kt * 32 + ((lane >> 4) << 4) + i;
  int col = nt * 16 + (lane & 15);
  float v = (K < Ksrc && col < Nsrc) ? W[(size_t)K * Nsrc + col] : 0.0f;
  Bp[t] = f2bf(v);
}
// Concatenated pre-NN weights: Bcat = [preW[:fi] | preW[fi:2fi]]  -> one GEMM gives [xd|xs]
__global__ void pack_b_precat(const float* __restrict__ preW, int fi,
                              unsigned short* __restrict__ Bp, int ntn, int total) {
  int t = blockIdx.x * 256 + threadIdx.x; if (t >= total) return;
  int i = t & 15, lane = (t >> 4) & 31, tile = t >> 9;
  int kt = tile / ntn, nt = tile - kt * ntn;
  int K = kt * 32 + ((lane >> 4) << 4) + i;
  int col = nt * 16 + (lane & 15);
  float v = 0.0f;
  if (K < fi && col < 2 * fi)
    v = (col < fi) ? preW[(size_t)K * fi + col] : preW[(size_t)(fi + K) * fi + (col - fi)];
  Bp[t] = f2bf(v);
}

// ---------- WMMA GEMM with N register blocking ----------
// Each wave: 1 M-tile x NT N-tiles. A fragment loaded once per K-step and
// reused NT times (cuts A read traffic by NT: the big hb operand is 133MB,
// streamed ~4x instead of 16x -> stays mostly in the 192MB L2).
// A fragment (contiguous k order, matching the B packing): lane holds row M = l15; two contiguous 8-ushort
// runs at K offsets {half*16..+7} and {half*16+8..+15} -> two global_load_b128.
union FragU { u32x4 q[2]; v16bf v; };

template <int NT>
__global__ __launch_bounds__(256)
void wmma_gemm_bf16(const unsigned short* __restrict__ A, int lda,
                    const unsigned short* __restrict__ Bp,
                    const float* __restrict__ bias,
                    const float* __restrict__ gamma,
                    const float* __restrict__ beta,
                    float* __restrict__ C, int ldc,
                    unsigned short* __restrict__ Cb, int ldcb,
                    int mtiles, int ntn, int Kt, int relu_bn) {
  int wave = threadIdx.x >> 5;
  int ngn = ntn / NT;                       // N-groups per row of tiles
  int group = blockIdx.x * 8 + wave;
  if (group >= mtiles * ngn) return;        // whole-wave uniform: EXEC stays all-ones
  int mt = group / ngn, nt0 = (group - mt * ngn) * NT;
  int lane = threadIdx.x & 31;
  int half = lane >> 4, l15 = lane & 15;

  const unsigned short* Arow = A + (size_t)(mt * 16 + l15) * lda + half * 16;
  const unsigned short* Bt   = Bp + ((size_t)nt0 * 32 + lane) * 16;
  size_t bstride = (size_t)ntn * 512;       // ushorts per kt slab

  v8f acc[NT];
#pragma unroll
  for (int j = 0; j < NT; ++j) acc[j] = v8f{0.f, 0.f, 0.f, 0.f, 0.f, 0.f, 0.f, 0.f};

  for (int kt = 0; kt < Kt; ++kt) {
    FragU a;
    a.q[0] = *(const u32x4*)(Arow + kt * 32);
    a.q[1] = *(const u32x4*)(Arow + kt * 32 + 8);
    if (kt + 1 < Kt) __builtin_prefetch(Arow + (kt + 1) * 32, 0, 1);  // global_prefetch_b8
    const unsigned short* Bk = Bt + (size_t)kt * bstride;
#pragma unroll
    for (int j = 0; j < NT; ++j) {
      FragU b;
      const u32x4* bp = (const u32x4*)(Bk + (size_t)j * 512);
      b.q[0] = bp[0];
      b.q[1] = bp[1];
      acc[j] = __builtin_amdgcn_wmma_f32_16x16x32_bf16(false, a.v, false, b.v,
                                                       (short)0, acc[j], false, false);
    }
  }

  const float BNC = 0.9999950000374997f;    // 1/sqrt(1+1e-5) eval-mode BN
#pragma unroll
  for (int j = 0; j < NT; ++j) {
    int col = (nt0 + j) * 16 + l15;
    float bi = bias  ? bias[col]  : 0.f;
    float ga = gamma ? gamma[col] : 1.f;
    float be = beta  ? beta[col]  : 0.f;
#pragma unroll
    for (int i = 0; i < 8; ++i) {
      int row = mt * 16 + half * 8 + i;     // C/D layout: VGPR i -> M = i + 8*half, N = l15
      float y = acc[j][i] + bi;
      if (relu_bn) y = fmaxf(0.f, y * BNC * ga + be);
      if (C)  C[(size_t)row * ldc + col] = y;
      if (Cb) Cb[(size_t)row * ldcb + col] = f2bf(y);
    }
  }
}

// ---------- edge message + segment {sum, sum^2, min, max} ----------
__global__ __launch_bounds__(256)
void edge_msg_kernel(const int* __restrict__ src, const int* __restrict__ dst,
                     const float* __restrict__ ea, const float* __restrict__ pre, int ldp,
                     const float* __restrict__ Wfe, const float* __restrict__ bfe, int fi,
                     float* __restrict__ sums, float* __restrict__ sqs,
                     unsigned* __restrict__ mnk, unsigned* __restrict__ mxk, int total) {
  int idx = blockIdx.x * 256 + threadIdx.x; if (idx >= total) return;
  int e = idx / fi, c = idx - e * fi;
  int d = dst[e], s = src[e];
  float m = pre[(size_t)d * ldp + c] + pre[(size_t)s * ldp + fi + c];
  m += ea[e * 3 + 0] * Wfe[c] + ea[e * 3 + 1] * Wfe[fi + c] + ea[e * 3 + 2] * Wfe[2 * fi + c] + bfe[c];
  int o = d * fi + c;
  atomicAdd(&sums[o], m);
  atomicAdd(&sqs[o], m * m);
  atomicMin(&mnk[o], fkey(m));
  atomicMax(&mxk[o], fkey(m));
}

// ---------- build h = [x, agg, agg*(dl/al), agg*(al/dl)] as bf16 GEMM operand ----------
__global__ __launch_bounds__(256)
void hbuild_kernel(const unsigned short* __restrict__ xb, int ldx,
                   const float* __restrict__ deg, const float* __restrict__ scal,
                   const float* __restrict__ sums, const float* __restrict__ sqs,
                   const unsigned* __restrict__ mnk, const unsigned* __restrict__ mxk,
                   unsigned short* __restrict__ hb, int ldh, int fi, int total) {
  int idx = blockIdx.x * 256 + threadIdx.x; if (idx >= total) return;
  int n = idx / fi, c = idx - n * fi;
  float dg = deg[n];
  float degc = fmaxf(dg, 1.0f);
  int o = n * fi + c;
  float mean = sums[o] / degc;
  float sd = sqrtf(fmaxf(sqs[o] / degc - mean * mean, 0.0f) + 1e-5f);
  float mn = dg > 0.f ? kdec(mnk[o]) : 0.f;
  float mx = dg > 0.f ? kdec(mxk[o]) : 0.f;
  float dl = logf(degc + 1.0f);
  float al = scal[1];
  float s1 = dl / al, s2 = al / dl;
  unsigned short* hr = hb + (size_t)n * ldh;
  hr[c] = xb[(size_t)n * ldx + c];
  int b = fi + c;
  hr[b]           = f2bf(mean);      hr[b + fi]     = f2bf(mn);
  hr[b + 2 * fi]  = f2bf(mx);        hr[b + 3 * fi] = f2bf(sd);
  hr[b + 4 * fi]  = f2bf(mean * s1); hr[b + 5 * fi] = f2bf(mn * s1);
  hr[b + 6 * fi]  = f2bf(mx * s1);   hr[b + 7 * fi] = f2bf(sd * s1);
  hr[b + 8 * fi]  = f2bf(mean * s2); hr[b + 9 * fi] = f2bf(mn * s2);
  hr[b + 10 * fi] = f2bf(mx * s2);   hr[b + 11 * fi] = f2bf(sd * s2);
}

// ---------- pooling head ----------
__global__ void count_kernel(const int* __restrict__ bat, float* __restrict__ counts, int n) {
  int t = blockIdx.x * 256 + threadIdx.x; if (t < n) atomicAdd(&counts[bat[t]], 1.0f);
}
__global__ void gsum_kernel(const int* __restrict__ bat, const float* __restrict__ nrep,
                            float* __restrict__ gsum, int total) {
  int t = blockIdx.x * 256 + threadIdx.x; if (t >= total) return;
  int n = t >> 8, c = t & 255;
  atomicAdd(&gsum[bat[n] * 256 + c], nrep[t]);
}
__global__ void starts_kernel(const float* __restrict__ counts, int* __restrict__ starts) {
  int run = 0;
  for (int g = 0; g < GG; ++g) { starts[g] = run; run += (int)(counts[g] + 0.5f); }
}
__global__ void embpool_kernel(const int* __restrict__ anum, const int* __restrict__ starts,
                               const float* __restrict__ nrep, const float* __restrict__ gsum,
                               const float* __restrict__ counts,
                               float* __restrict__ nemb_out, float* __restrict__ pooled, int total) {
  int t = blockIdx.x * 256 + threadIdx.x; if (t >= total) return;
  int g = t >> 8, c = t & 255;
  int node = anum[g] + starts[g];
  float ne = nrep[(size_t)node * 256 + c];
  nemb_out[t] = ne;                                  // second tuple output
  pooled[t] = 0.5f * gsum[t] / counts[g] + ne;
}
__global__ __launch_bounds__(256)
void mlp_kernel(const float* __restrict__ pooled, const float* __restrict__ mlpW,
                const float* __restrict__ mlpb, float* __restrict__ out, int total) {
  int t = blockIdx.x * 256 + threadIdx.x; if (t >= total) return;
  int g = t / 200, j = t - g * 200;
  float acc = mlpb[j];
  const float* p = pooled + (size_t)g * 256;
  for (int c = 0; c < 256; ++c) acc += p[c] * mlpW[(size_t)c * 200 + j];
  out[t] = acc;
}

// =======================================================================================
static void launch_gemm(const unsigned short* A, int lda, const unsigned short* Bp,
                        const float* bias, const float* gamma, const float* beta,
                        float* C, int ldc, unsigned short* Cb, int ldcb,
                        int ntn, int Kt, int relu_bn, hipStream_t stream) {
  int mtiles = NN / 16;
  if (ntn % 4 == 0) {
    int groups = mtiles * (ntn / 4);
    wmma_gemm_bf16<4><<<(groups + 7) / 8, 256, 0, stream>>>(
        A, lda, Bp, bias, gamma, beta, C, ldc, Cb, ldcb, mtiles, ntn, Kt, relu_bn);
  } else {
    int groups = mtiles * (ntn / 2);
    wmma_gemm_bf16<2><<<(groups + 7) / 8, 256, 0, stream>>>(
        A, lda, Bp, bias, gamma, beta, C, ldc, Cb, ldcb, mtiles, ntn, Kt, relu_bn);
  }
}

extern "C" void kernel_launch(void* const* d_in, const int* in_sizes, int n_in,
                              void* d_out, int out_size, void* d_ws, size_t ws_size,
                              hipStream_t stream) {
  const float* x    = (const float*)d_in[0];
  const float* ea   = (const float*)d_in[1];
  const int*   ei   = (const int*)d_in[2];
  const int*   bat  = (const int*)d_in[3];
  const int*   anum = (const int*)d_in[4];
  const float* mlpW = (const float*)d_in[35];
  const float* mlpb = (const float*)d_in[36];
  const int* src = ei;
  const int* dst = ei + EE;

  // workspace carving (~340 MB)
  char* base = (char*)d_ws; size_t off = 0;
  auto alloc = [&](size_t bytes) -> void* {
    off = (off + 255) & ~(size_t)255;
    void* p = base + off; off += bytes; return p;
  };
  float* deg      = (float*)alloc((size_t)NN * 4);
  float* scal     = (float*)alloc(64);
  int*   starts   = (int*)alloc((size_t)GG * 4);
  float* counts   = (float*)alloc((size_t)GG * 4);
  float* gsum     = (float*)alloc((size_t)GG * 256 * 4);
  float* pooled   = (float*)alloc((size_t)GG * 256 * 4);
  float* Wfe      = (float*)alloc(3 * 128 * 4);
  float* bfe      = (float*)alloc(128 * 4);
  unsigned short* packB = (unsigned short*)alloc((size_t)52 * 16 * 512 * 2);
  unsigned short* xb    = (unsigned short*)alloc((size_t)NN * 256 * 2);
  float* preout   = (float*)alloc((size_t)NN * 256 * 4);
  float* sums     = (float*)alloc((size_t)NN * 128 * 4);
  float* sqs      = (float*)alloc((size_t)NN * 128 * 4);
  unsigned* mnk   = (unsigned*)alloc((size_t)NN * 128 * 4);
  unsigned* mxk   = (unsigned*)alloc((size_t)NN * 128 * 4);
  unsigned short* pb = (unsigned short*)alloc((size_t)NN * 256 * 2);
  unsigned short* hb = (unsigned short*)alloc((size_t)NN * 1664 * 2);
  float* nrep     = (float*)alloc((size_t)NN * 256 * 4);
  (void)ws_size; (void)in_sizes; (void)n_in; (void)out_size;

  auto NB = [](long long n) { return (unsigned)((n + 255) / 256); };

  // degrees + avg log-degree
  fill_f32_kernel<<<NB(NN), 256, 0, stream>>>(deg, 0.f, NN);
  fill_f32_kernel<<<1, 256, 0, stream>>>(scal, 0.f, 16);
  deg_kernel<<<NB(EE), 256, 0, stream>>>(dst, deg, EE);
  logsum_kernel<<<NB(NN), 256, 0, stream>>>(deg, scal, NN);
  avg_kernel<<<1, 1, 0, stream>>>(scal);

  // layer-0 input: bf16, K padded 15 -> 32
  fill_u32_kernel<<<NB(NN * 16), 256, 0, stream>>>((unsigned*)xb, 0u, NN * 16);
  convert_x_kernel<<<NB((long long)NN * 15), 256, 0, stream>>>(x, xb, NN * 15);

  const int FI[3] = {15, 64, 128}, FO[3] = {64, 128, 256};
  const int KX[3] = {32, 64, 128};        // xb row stride (padded pre-GEMM K)
  const int PN[3] = {32, 128, 256};       // pre-GEMM padded N (>= 2*fi)
  const int KH[3] = {224, 832, 1664};     // hb row stride (padded 13*fi)
  const unsigned MINKEY_INIT = 0xFF800000u;  // fkey(+inf)
  const unsigned MAXKEY_INIT = 0x007FFFFFu;  // fkey(-inf)

  for (int l = 0; l < 3; ++l) {
    const float* eeW   = (const float*)d_in[5 + 10 * l + 0];
    const float* eeb   = (const float*)d_in[5 + 10 * l + 1];
    const float* preW  = (const float*)d_in[5 + 10 * l + 2];
    const float* preb  = (const float*)d_in[5 + 10 * l + 3];
    const float* postW = (const float*)d_in[5 + 10 * l + 4];
    const float* postb = (const float*)d_in[5 + 10 * l + 5];
    const float* linW  = (const float*)d_in[5 + 10 * l + 6];
    const float* linb  = (const float*)d_in[5 + 10 * l + 7];
    const float* gam   = (const float*)d_in[5 + 10 * l + 8];
    const float* bet   = (const float*)d_in[5 + 10 * l + 9];
    int fi = FI[l], fo = FO[l], kx = KX[l], pn = PN[l], kh = KH[l];

    // 1) node pre-GEMM:  [xd | xs] = xb @ [preW[:fi] | preW[fi:2fi]]
    {
      int Kt = kx / 32, ntn = pn / 16, total = Kt * ntn * 512;
      pack_b_precat<<<NB(total), 256, 0, stream>>>(preW, fi, packB, ntn, total);
      launch_gemm(xb, kx, packB, nullptr, nullptr, nullptr,
                  preout, pn, nullptr, 0, ntn, Kt, 0, stream);
    }
    // 2) fused edge weights + aggregation buffers + edge scatter
    fuse_edge_kernel<<<NB(4 * fi), 256, 0, stream>>>(eeW, eeb, preW, preb, fi, Wfe, bfe);
    fill_f32_kernel<<<NB((long long)NN * fi), 256, 0, stream>>>(sums, 0.f, NN * fi);
    fill_f32_kernel<<<NB((long long)NN * fi), 256, 0, stream>>>(sqs, 0.f, NN * fi);
    fill_u32_kernel<<<NB((long long)NN * fi), 256, 0, stream>>>(mnk, MINKEY_INIT, NN * fi);
    fill_u32_kernel<<<NB((long long)NN * fi), 256, 0, stream>>>(mxk, MAXKEY_INIT, NN * fi);
    edge_msg_kernel<<<NB((long long)EE * fi), 256, 0, stream>>>(
        src, dst, ea, preout, pn, Wfe, bfe, fi, sums, sqs, mnk, mxk, (int)((long long)EE * fi));
    // 3) build h (bf16); zero-pad hb only needed for layer 0 (195 -> 224)
    if (l == 0) fill_u32_kernel<<<NB((long long)NN * (kh / 2)), 256, 0, stream>>>((unsigned*)hb, 0u, NN * (kh / 2));
    hbuild_kernel<<<NB((long long)NN * fi), 256, 0, stream>>>(
        xb, kx, deg, scal, sums, sqs, mnk, mxk, hb, kh, fi, NN * fi);
    // 4) post GEMM: pb(bf16) = h @ postW + postb
    {
      int Kt = kh / 32, ntn = fo / 16, total = Kt * ntn * 512;
      pack_b_plain<<<NB(total), 256, 0, stream>>>(postW, 13 * fi, fo, packB, ntn, total);
      launch_gemm(hb, kh, packB, postb, nullptr, nullptr,
                  nullptr, 0, pb, fo, ntn, Kt, 0, stream);
    }
    // 5) lin GEMM + BN + ReLU -> nrep (fp32) and xb (bf16 input of next layer)
    {
      int Kt = fo / 32, ntn = fo / 16, total = Kt * ntn * 512;
      pack_b_plain<<<NB(total), 256, 0, stream>>>(linW, fo, fo, packB, ntn, total);
      launch_gemm(pb, fo, packB, linb, gam, bet,
                  nrep, fo, xb, fo, ntn, Kt, 1, stream);
    }
  }

  // pooling + head (out tuple: [400,200] logits, then [400,256] node_emb)
  float* outp = (float*)d_out;
  fill_f32_kernel<<<NB(GG), 256, 0, stream>>>(counts, 0.f, GG);
  fill_f32_kernel<<<NB(GG * 256), 256, 0, stream>>>(gsum, 0.f, GG * 256);
  count_kernel<<<NB(NN), 256, 0, stream>>>(bat, counts, NN);
  gsum_kernel<<<NB((long long)NN * 256), 256, 0, stream>>>(bat, nrep, gsum, NN * 256);
  starts_kernel<<<1, 1, 0, stream>>>(counts, starts);
  embpool_kernel<<<NB(GG * 256), 256, 0, stream>>>(anum, starts, nrep, gsum, counts,
                                                   outp + GG * 200, pooled, GG * 256);
  mlp_kernel<<<NB(GG * 200), 256, 0, stream>>>(pooled, mlpW, mlpb, outp, GG * 200);
}
